// FullAttention_46093589021349
// MI455X (gfx1250) — matrix operand order repaired and checked by reading, not compiled, not run on hardware
//
#include <hip/hip_runtime.h>

typedef __attribute__((ext_vector_type(16))) _Float16 v16h;
typedef __attribute__((ext_vector_type(8)))  _Float16 v8h;
typedef __attribute__((ext_vector_type(2)))  _Float16 v2h;
typedef __attribute__((ext_vector_type(8)))  float    v8f;
typedef __attribute__((ext_vector_type(4)))  float    v4f;
typedef __attribute__((ext_vector_type(8)))  int      v8i;

namespace {
constexpr int CB = 4;      // batch
constexpr int CL = 2048;   // query length
constexpr int CH = 16;     // heads
constexpr int CE = 64;     // qk dim
constexpr int CS = 2048;   // key length
constexpr int CD = 64;     // v dim
// Base-2 softmax: fold 1/sqrt(64) * log2(e) into Q so exp2 is a bare v_exp_f32.
constexpr float CQSCALE = 0.125f * 1.44269504088896340736f;

constexpr int WAVES = 8;             // waves per block
constexpr int ROWS_PER_BLOCK = 128;  // 8 waves * 16 query rows

struct MaskOff { static constexpr bool value = false; };
struct MaskOn  { static constexpr bool value = true;  };
}

// ---------------------------------------------------------------------------
// Pass 1a: K [B,S,H,E] f32  ->  Kh [B,H,S,E] f16   (row remap, coalesced)
// ---------------------------------------------------------------------------
__global__ __launch_bounds__(256)
void convert_k_f16(const float* __restrict__ K, _Float16* __restrict__ Kh)
{
    const int tid = blockIdx.x * 256 + threadIdx.x;   // B*S*H*32 threads
    const int e2  = tid & 31;                         // f16 pair within row
    const int row = tid >> 5;                         // (b,s,h) row
    const int h   = row % CH;
    const int s   = (row / CH) % CS;
    const int b   = row / (CH * CS);

    const float* src = K + (size_t)row * CE + e2 * 2;
    _Float16* dst = Kh + (((size_t)(b * CH + h) * CS + s) * CE) + e2 * 2;
    v2h p; p[0] = (_Float16)src[0]; p[1] = (_Float16)src[1];
    *(v2h*)dst = p;
}

// ---------------------------------------------------------------------------
// Pass 1b: V [B,S,H,D] f32  ->  Vt [B,H,D,S] f16   (LDS-tiled transpose)
// ---------------------------------------------------------------------------
__global__ __launch_bounds__(256)
void transpose_v_f16(const float* __restrict__ V, _Float16* __restrict__ Vt)
{
    __shared__ float tile[64][65];  // 64 s-rows x 64 d-cols, padded

    const int st = blockIdx.x % (CS / 64);
    const int bh = blockIdx.x / (CS / 64);
    const int h  = bh % CH;
    const int b  = bh / CH;
    const int t  = threadIdx.x;

    {   // coalesced read: thread -> (s-row, 16 consecutive d)
        const int sr = t >> 2;
        const int c0 = (t & 3) * 16;
        const float* src = V + (((size_t)b * CS + st * 64 + sr) * CH + h) * CD + c0;
#pragma unroll
        for (int i = 0; i < 16; ++i) tile[sr][c0 + i] = src[i];
    }
    __syncthreads();
    {   // write transposed: thread -> (d-row, 16 consecutive s), packed f16x2
        const int dr = t >> 2;
        const int s0 = (t & 3) * 16;
        _Float16* dst = Vt + ((size_t)(b * CH + h) * CD + dr) * CS + st * 64 + s0;
#pragma unroll
        for (int i = 0; i < 16; i += 2) {
            v2h p;
            p[0] = (_Float16)tile[s0 + i][dr];
            p[1] = (_Float16)tile[s0 + i + 1][dr];
            *(v2h*)(dst + i) = p;
        }
    }
}

// ---------------------------------------------------------------------------
// Pass 2: flash attention, transposed-score formulation.
//   S^T = K @ Q^T   (A = K tile 16x32, B = Q^T 32x16) -> lane owns ONE query col
//   O^T = V^T @ P^T (A = V^T chunk 16x32, B = P^T 32x16)
// Softmax over keys: in-lane tree + one shfl_xor(16). Mask-free main loop,
// masked diagonal tail (<=2 blocks per wave). Base-2 softmax with raw
// v_exp_f32 (__builtin_amdgcn_exp2f) -- denormal flush is fine for softmax.
// ---------------------------------------------------------------------------
template<bool USE_F16>
__global__ __launch_bounds__(256)
void fa_fwd_causal_wmma(const float* __restrict__ Q,
                        const float* __restrict__ K,
                        const float* __restrict__ V,
                        const _Float16* __restrict__ Kh,
                        const _Float16* __restrict__ Vt,
                        float* __restrict__ O)
{
    __shared__ __align__(16) float obuf[WAVES][16][68];  // epilogue transpose

    const int lane = threadIdx.x & 31;
    const int wv   = threadIdx.x >> 5;
    const int hfw  = lane >> 4;   // 16-lane half
    const int l16  = lane & 15;

    const int tiles_per_bh = CL / ROWS_PER_BLOCK;
    const int tile = blockIdx.x % tiles_per_bh;
    const int bh   = blockIdx.x / tiles_per_bh;
    const int h    = bh % CH;
    const int b    = bh / CH;

    const int qw = tile * ROWS_PER_BLOCK + wv * 16;  // first query row of wave
    const int qi = qw + l16;                         // this lane's query column

    // ---- Q^T B-fragments (loaded once, scale*log2e folded in).
    v16h bq[2];
    {
        const float* qrow = Q + (((size_t)b * CL + qi) * CH + h) * CE;
#pragma unroll
        for (int c = 0; c < 2; ++c) {
            const int e0 = 32 * c + 8 * hfw;
#pragma unroll
            for (int i = 0; i < 16; ++i)
                bq[c][i] = (_Float16)(qrow[e0 + (i < 8 ? i : 8 + i)] * CQSCALE);
        }
    }

    v8f acc[4] = {};                 // O^T tiles: rows d=16t.., col = lane query
    float mrun = -__builtin_inff();  // per-lane (per-query) softmax state, log2
    float lrun = 0.0f;

    const _Float16* khb = Kh + (size_t)(b * CH + h) * CS * CE;
    const _Float16* vtb = Vt + (size_t)(b * CH + h) * CD * CS;
    const float* kfb = K + (((size_t)b * CS) * CH + h) * CE;
    const float* vfb = V + (((size_t)b * CS) * CH + h) * CD;

    const int nblocks = (qw + 47) >> 5;  // causal: keys [0, qw+16), 32/iter
    const int nfree   = (qw + 1)  >> 5;  // blocks entirely below the diagonal

    auto body = [&](int k0, auto mask_tag) {
        constexpr bool MASKED = decltype(mask_tag)::value;

        // ---- K A-fragments: aK[s][c] = keys [k0+16s,+16) x E-chunk c.
        v16h aK[2][2];
#pragma unroll
        for (int s = 0; s < 2; ++s) {
            if constexpr (USE_F16) {
                const _Float16* kr = khb + (size_t)(k0 + 16 * s + l16) * CE;
#pragma unroll
                for (int c = 0; c < 2; ++c) {
                    const v8h lo = *(const v8h*)(kr + 32 * c + 8 * hfw);
                    const v8h hi = *(const v8h*)(kr + 32 * c + 16 + 8 * hfw);
#pragma unroll
                    for (int i = 0; i < 8; ++i) { aK[s][c][i] = lo[i]; aK[s][c][8 + i] = hi[i]; }
                }
            } else {
                const float* kr = kfb + (size_t)(k0 + 16 * s + l16) * (CH * CE);
#pragma unroll
                for (int c = 0; c < 2; ++c) {
#pragma unroll
                    for (int i = 0; i < 8; ++i) {
                        aK[s][c][i]     = (_Float16)kr[32 * c + 8 * hfw + i];
                        aK[s][c][8 + i] = (_Float16)kr[32 * c + 16 + 8 * hfw + i];
                    }
                }
            }
        }

        // ---- Scores S^T (log2-scaled): rows = keys, cols = queries.
        v8f sc0 = {}, sc1 = {};
        sc0 = __builtin_amdgcn_wmma_f32_16x16x32_f16(false, aK[0][0], false, bq[0], (short)0, sc0, false, false);
        sc0 = __builtin_amdgcn_wmma_f32_16x16x32_f16(false, aK[0][1], false, bq[1], (short)0, sc0, false, false);
        sc1 = __builtin_amdgcn_wmma_f32_16x16x32_f16(false, aK[1][0], false, bq[0], (short)0, sc1, false, false);
        sc1 = __builtin_amdgcn_wmma_f32_16x16x32_f16(false, aK[1][1], false, bq[1], (short)0, sc1, false, false);

        // ---- (Optional) causal mask + block max. Element r of sc0 is key
        // k0 + r + 8*hfw; of sc1 is key k0 + 16 + r + 8*hfw.
        float p0[8], p1[8];
        float mblk = -__builtin_inff();
        if constexpr (MASKED) {
            const int th = qi - k0 - 8 * hfw;  // sc0[r] visible iff r <= th
#pragma unroll
            for (int r = 0; r < 8; ++r) {
                p0[r] = (r <= th)      ? sc0[r] : -__builtin_inff();
                p1[r] = (r + 16 <= th) ? sc1[r] : -__builtin_inff();
                mblk = fmaxf(mblk, fmaxf(p0[r], p1[r]));
            }
        } else {
#pragma unroll
            for (int r = 0; r < 8; ++r) {
                p0[r] = sc0[r];
                p1[r] = sc1[r];
                mblk = fmaxf(mblk, fmaxf(p0[r], p1[r]));
            }
        }
        mblk = fmaxf(mblk, __shfl_xor(mblk, 16, 32));

        // ---- Online softmax in base 2: one raw v_exp_f32 per element.
        const float mnew  = fmaxf(mrun, mblk);
        const float alpha = __builtin_amdgcn_exp2f(mrun - mnew);
        mrun = mnew;
        float psum = 0.0f;
#pragma unroll
        for (int r = 0; r < 8; ++r) {
            p0[r] = __builtin_amdgcn_exp2f(p0[r] - mnew);
            p1[r] = __builtin_amdgcn_exp2f(p1[r] - mnew);
            psum += p0[r] + p1[r];
        }
        psum += __shfl_xor(psum, 16, 32);
        lrun = lrun * alpha + psum;
#pragma unroll
        for (int t = 0; t < 4; ++t) acc[t] = acc[t] * alpha;  // uniform per lane

        // ---- Build P^T B-fragment. Each lane uses its own pk0/pk1: keys sit
        // in the same k order as aV, so the rcv from the partner is unused.
        // Pre-select what to SEND -> only 4 shuffles.
        int pk0[4], pk1[4], snd[4], rcv[4];
#pragma unroll
        for (int d = 0; d < 4; ++d) {
            v2h a; a[0] = (_Float16)p0[2 * d]; a[1] = (_Float16)p0[2 * d + 1];
            v2h c; c[0] = (_Float16)p1[2 * d]; c[1] = (_Float16)p1[2 * d + 1];
            pk0[d] = __builtin_bit_cast(int, a);
            pk1[d] = __builtin_bit_cast(int, c);
            snd[d] = hfw ? pk0[d] : pk1[d];    // sent to partner; its rcv is unused
        }
#pragma unroll
        for (int d = 0; d < 4; ++d) rcv[d] = __shfl_xor(snd[d], 16, 32);
        v8i bwi;
#pragma unroll
        for (int d = 0; d < 4; ++d) {
            bwi[d]     = pk0[d];  // keys {k0..7} / {k0+8..15}
            bwi[4 + d] = pk1[d];  // keys {k0+16..23} / {k0+24..31}
        }
        const v16h bp = __builtin_bit_cast(v16h, bwi);

        // ---- O^T += V^T @ P^T over 4 D-row chunks.
#pragma unroll
        for (int t = 0; t < 4; ++t) {
            v16h aV;
            if constexpr (USE_F16) {
                const _Float16* vr = vtb + (size_t)(16 * t + l16) * CS + k0;
                const v8h lo = *(const v8h*)(vr + 8 * hfw);
                const v8h hi = *(const v8h*)(vr + 16 + 8 * hfw);
#pragma unroll
                for (int i = 0; i < 8; ++i) { aV[i] = lo[i]; aV[8 + i] = hi[i]; }
            } else {
                const float* vc = vfb + (16 * t + l16);
#pragma unroll
                for (int i = 0; i < 8; ++i) {
                    aV[i]     = (_Float16)vc[(size_t)(k0 + 8 * hfw + i) * (CH * CD)];
                    aV[8 + i] = (_Float16)vc[(size_t)(k0 + 16 + 8 * hfw + i) * (CH * CD)];
                }
            }
            acc[t] = __builtin_amdgcn_wmma_f32_16x16x32_f16(false, aV, false, bp, (short)0, acc[t], false, false);
        }
    };

    int j = 0;
    for (; j < nfree; ++j)   body(j << 5, MaskOff{});  // mask-free main loop
    for (; j < nblocks; ++j) body(j << 5, MaskOn{});   // diagonal tail (<=2)

    // ---- Epilogue: divide by l, transpose O^T -> O through wave-local LDS.
    const float inv = 1.0f / lrun;
#pragma unroll
    for (int t = 0; t < 4; ++t) acc[t] = acc[t] * inv;

    {
        float* orow = &obuf[wv][l16][0];
#pragma unroll
        for (int t = 0; t < 4; ++t) {
            v4f lo, hi;
#pragma unroll
            for (int i = 0; i < 4; ++i) { lo[i] = acc[t][i]; hi[i] = acc[t][4 + i]; }
            *(v4f*)(orow + 16 * t + 8 * hfw)     = lo;  // d = 16t+8hfw+0..3
            *(v4f*)(orow + 16 * t + 8 * hfw + 4) = hi;  // d = 16t+8hfw+4..7
        }
    }
    __builtin_amdgcn_wave_barrier();  // wave-local LDS ordering
    {
        const float* rrow = &obuf[wv][l16][hfw * 32];
        float* og = O + (((size_t)b * CL + (qw + l16)) * CH + h) * CD + hfw * 32;
#pragma unroll
        for (int i = 0; i < 8; ++i)
            ((v4f*)og)[i] = ((const v4f*)rrow)[i];
    }
}

extern "C" void kernel_launch(void* const* d_in, const int* in_sizes, int n_in,
                              void* d_out, int out_size, void* d_ws, size_t ws_size,
                              hipStream_t stream) {
    (void)in_sizes; (void)n_in; (void)out_size;
    const float* Q = (const float*)d_in[0];
    const float* K = (const float*)d_in[1];
    const float* V = (const float*)d_in[2];
    float* O = (float*)d_out;

    const dim3 grid(CB * CH * (CL / ROWS_PER_BLOCK));  // 1024 workgroups
    const dim3 block(256);                              // 8 wave32 waves

    const size_t kh_elems = (size_t)CB * CH * CS * CE;
    const size_t vt_elems = (size_t)CB * CH * CD * CS;
    const size_t need = (kh_elems + vt_elems) * sizeof(_Float16);  // 32 MB

    if (ws_size >= need) {
        _Float16* Kh = (_Float16*)d_ws;
        _Float16* Vt = Kh + kh_elems;
        convert_k_f16<<<dim3((CB * CS * CH * 32) / 256), dim3(256), 0, stream>>>(K, Kh);
        transpose_v_f16<<<dim3(CB * CH * (CS / 64)), dim3(256), 0, stream>>>(V, Vt);
        fa_fwd_causal_wmma<true><<<grid, block, 0, stream>>>(Q, K, V, Kh, Vt, O);
    } else {
        fa_fwd_causal_wmma<false><<<grid, block, 0, stream>>>(Q, K, V, nullptr, nullptr, O);
    }
}
